// GraphTransformerLayer_7464653161131
// MI455X (gfx1250) — matrix operand order repaired and checked by reading, not compiled, not run on hardware
//
#include <hip/hip_runtime.h>
#include <hip/hip_bf16.h>
#include <math.h>

typedef __attribute__((ext_vector_type(16))) _Float16 v16h;
typedef __attribute__((ext_vector_type(8)))  float    v8f;

#define C_IN 32
#define HD 128
#define QKV_STRIDE 384   // q(128) | k(128) | v(128)

// ---------------------------------------------------------------------------
// Fold the two edge matrices: Wc[o][d] = sum_c W_lin_edge[o][c] * W_edge_emb[c][d]
// ---------------------------------------------------------------------------
__global__ void wc_kernel(const float* __restrict__ Wle,
                          const float* __restrict__ Wee,
                          float* __restrict__ Wc) {
  int t = threadIdx.x;            // 256 threads = 128 x 2
  int o = t >> 1, d = t & 1;
  float s = 0.f;
#pragma unroll
  for (int c = 0; c < C_IN; ++c) s = fmaf(Wle[o * C_IN + c], Wee[c * 2 + d], s);
  Wc[o * 2 + d] = s;
}

// ---------------------------------------------------------------------------
// One weight's 8 column-tiles for a 16-node row tile.  STRIDE is compile-time
// so row stores fold into base + immediate-offset store clauses.
// ---------------------------------------------------------------------------
template <int STRIDE>
__device__ __forceinline__ void
proj_store_tiles(v16h a, const float* __restrict__ W, float* __restrict__ dst,
                 int mtile, int m, int half, bool full, int N) {
  const int rowbase = mtile * 16 + (half << 3);   // D row = reg + 8*half
#pragma unroll
  for (int ti = 0; ti < 8; ++ti) {
    const int ocol = ti * 16 + m;                 // B column = lane&15
    const float* wr = W + ocol * C_IN;            // B[k][n] = W[n][k]
    v16h b;
#pragma unroll
    for (int v = 0; v < 8; ++v) {
      int k = (half << 4) + (v << 1);
      b[2 * v]     = (_Float16)wr[k];
      b[2 * v + 1] = (_Float16)wr[k + 1];
    }
    v8f c = {};
    c = __builtin_amdgcn_wmma_f32_16x16x32_f16(false, a, false, b, (short)0, c,
                                               false, false);
    float* p = dst + (size_t)rowbase * STRIDE + ti * 16 + m;
    if (full) {
#pragma unroll
      for (int r = 0; r < 8; ++r) p[(size_t)r * STRIDE] = c[r];
    } else {
#pragma unroll
      for (int r = 0; r < 8; ++r)
        if (rowbase + r < N) p[(size_t)r * STRIDE] = c[r];
    }
  }
}

// ---------------------------------------------------------------------------
// Node projections via WMMA: q,k,v -> qkv buffer; skip -> agg buffer (so the
// edge aggregation accumulates on top of the skip connection).
// Block = 128 threads = 4 waves; wave w computes weight w for one 16-node tile.
// ---------------------------------------------------------------------------
__global__ void __launch_bounds__(128)
proj_kernel(const float* __restrict__ x,
            const float* __restrict__ Wq, const float* __restrict__ Wk,
            const float* __restrict__ Wv, const float* __restrict__ Wsk,
            float* __restrict__ qkv, float* __restrict__ agg, int N) {
  const int wave = threadIdx.x >> 5;
  const int lane = threadIdx.x & 31;
  const int m    = lane & 15;
  const int half = lane >> 4;
  const int mtile = blockIdx.x;
  const bool full = (mtile * 16 + 16) <= N;       // block-uniform

  int arow = mtile * 16 + m;
  if (arow >= N) arow = N - 1;                    // clamp loads only
  const float* xr = x + (size_t)arow * C_IN;

  // A tile: 16x32 f32 -> f16, contiguous K order (k = 16*half + s), matching B
  v16h a;
#pragma unroll
  for (int v = 0; v < 8; ++v) {
    int k = (half << 4) + (v << 1);
    a[2 * v]     = (_Float16)xr[k];
    a[2 * v + 1] = (_Float16)xr[k + 1];
  }

  // Wave-uniform branch: EXEC stays all-ones inside each path.
  if (wave == 0)
    proj_store_tiles<QKV_STRIDE>(a, Wq, qkv, mtile, m, half, full, N);
  else if (wave == 1)
    proj_store_tiles<QKV_STRIDE>(a, Wk, qkv + HD, mtile, m, half, full, N);
  else if (wave == 2)
    proj_store_tiles<QKV_STRIDE>(a, Wv, qkv + 2 * HD, mtile, m, half, full, N);
  else
    proj_store_tiles<HD>(a, Wsk, agg, mtile, m, half, full, N);
}

// ---------------------------------------------------------------------------
// Per-(edge,head) attention logit + segment max (order-preserving uint atomics)
// ---------------------------------------------------------------------------
__global__ void edge_logits_kernel(const float* __restrict__ qkv,
                                   const float* __restrict__ ea,
                                   const float* __restrict__ Wc,
                                   const int* __restrict__ ei,
                                   float* __restrict__ logits,
                                   unsigned* __restrict__ segmax, int E) {
  int gid = blockIdx.x * blockDim.x + threadIdx.x;
  if (gid >= E * 4) return;
  int e = gid >> 2, h = gid & 3;
  int src = ei[e], dst = ei[E + e];
  float e0 = ea[e * 2], e1 = ea[e * 2 + 1];
  const float* qr = qkv + (size_t)dst * QKV_STRIDE + h * 32;       // q[dst]
  const float* kr = qkv + (size_t)src * QKV_STRIDE + HD + h * 32;  // k[src]
  const float* wc = Wc + h * 64;
  float acc = 0.f;
#pragma unroll
  for (int j = 0; j < 32; ++j) {
    float ke = fmaf(e0, wc[2 * j], fmaf(e1, wc[2 * j + 1], kr[j]));
    acc = fmaf(qr[j], ke, acc);
  }
  acc *= 0.17677669529663687f;  // 1/sqrt(32)
  logits[gid] = acc;
  unsigned fl  = __float_as_uint(acc);
  unsigned key = (fl & 0x80000000u) ? ~fl : (fl | 0x80000000u);
  atomicMax(segmax + dst * 4 + h, key);
}

// ---------------------------------------------------------------------------
// p = exp(logit - segmax[dst]); segment sum
// ---------------------------------------------------------------------------
__global__ void edge_softmax_kernel(const int* __restrict__ ei,
                                    const unsigned* __restrict__ segmax,
                                    float* __restrict__ p,
                                    float* __restrict__ segsum, int E) {
  int gid = blockIdx.x * blockDim.x + threadIdx.x;
  if (gid >= E * 4) return;
  int e = gid >> 2, h = gid & 3;
  int dst = ei[E + e];
  unsigned key = segmax[dst * 4 + h];
  unsigned fl  = (key & 0x80000000u) ? (key & 0x7FFFFFFFu) : ~key;
  float mx = __uint_as_float(fl);
  float pv = expf(p[gid] - mx);
  p[gid] = pv;
  atomicAdd(segsum + dst * 4 + h, pv);
}

// ---------------------------------------------------------------------------
// Weighted message aggregation: 32 lanes per edge, scatter-add into agg
// (agg already holds x @ W_skip.T)
// ---------------------------------------------------------------------------
__global__ void edge_aggregate_kernel(const float* __restrict__ qkv,
                                      const float* __restrict__ ea,
                                      const float* __restrict__ Wc,
                                      const int* __restrict__ ei,
                                      const float* __restrict__ p,
                                      const float* __restrict__ segsum,
                                      float* __restrict__ agg, int E) {
  long tid = (long)blockIdx.x * blockDim.x + threadIdx.x;
  if (tid >= (long)E * 32) return;
  int e = (int)(tid >> 5), lane = (int)(tid & 31);
  int src = ei[e], dst = ei[E + e];
  float e0 = ea[e * 2], e1 = ea[e * 2 + 1];
  const float* vr = qkv + (size_t)src * QKV_STRIDE + 2 * HD;  // v[src]
  float* ag = agg + (size_t)dst * HD;
#pragma unroll
  for (int h = 0; h < 4; ++h) {
    float alpha = p[e * 4 + h] / (segsum[dst * 4 + h] + 1e-16f);
    int o = h * 32 + lane;
    float msg = fmaf(e0, Wc[2 * o], fmaf(e1, Wc[2 * o + 1], vr[o])) * alpha;
    atomicAdd(ag + o, msg);
  }
}

// ---------------------------------------------------------------------------
// out = RMSNorm(agg @ W_out.T + x) * ln_w   via WMMA (K=128 -> 4 steps, 2 col tiles)
// Block = 128 threads = 4 waves, one 16-node tile per wave.
// ---------------------------------------------------------------------------
__global__ void __launch_bounds__(128)
out_kernel(const float* __restrict__ agg, const float* __restrict__ x,
           const float* __restrict__ Wout, const float* __restrict__ lnw,
           float* __restrict__ out, int N) {
  const int wave = threadIdx.x >> 5;
  const int lane = threadIdx.x & 31;
  const int m    = lane & 15;
  const int half = lane >> 4;
  const int ntiles = (N + 15) >> 4;
  const int tile = blockIdx.x * 4 + wave;
  if (tile >= ntiles) return;                 // wave-uniform: EXEC stays full
  const bool full = (tile * 16 + 16) <= N;    // wave-uniform

  int arow = tile * 16 + m;
  if (arow >= N) arow = N - 1;
  const float* ar = agg + (size_t)arow * HD;

  v16h a[4];
#pragma unroll
  for (int kt = 0; kt < 4; ++kt)
#pragma unroll
    for (int v = 0; v < 8; ++v) {
      int k = kt * 32 + (half << 4) + (v << 1);
      a[kt][2 * v]     = (_Float16)ar[k];
      a[kt][2 * v + 1] = (_Float16)ar[k + 1];
    }

  v8f c0 = {}, c1 = {};
#pragma unroll
  for (int kt = 0; kt < 4; ++kt) {            // output channels 0..15
    const float* wr = Wout + (size_t)m * HD + kt * 32 + (half << 4);
    v16h b;
#pragma unroll
    for (int v = 0; v < 8; ++v) {
      b[2 * v] = (_Float16)wr[2 * v]; b[2 * v + 1] = (_Float16)wr[2 * v + 1];
    }
    c0 = __builtin_amdgcn_wmma_f32_16x16x32_f16(false, a[kt], false, b, (short)0,
                                                c0, false, false);
  }
#pragma unroll
  for (int kt = 0; kt < 4; ++kt) {            // output channels 16..31
    const float* wr = Wout + (size_t)(16 + m) * HD + kt * 32 + (half << 4);
    v16h b;
#pragma unroll
    for (int v = 0; v < 8; ++v) {
      b[2 * v] = (_Float16)wr[2 * v]; b[2 * v + 1] = (_Float16)wr[2 * v + 1];
    }
    c1 = __builtin_amdgcn_wmma_f32_16x16x32_f16(false, a[kt], false, b, (short)0,
                                                c1, false, false);
  }

  const float lw0 = lnw[m], lw1 = lnw[16 + m];
  const int rowbase = tile * 16 + (half << 3);
  const float* xp = x + (size_t)rowbase * C_IN;
  float* op = out + (size_t)rowbase * C_IN;

  if (full) {
#pragma unroll
    for (int r = 0; r < 8; ++r) {
      float v0 = c0[r] + xp[(size_t)r * C_IN + m];
      float v1 = c1[r] + xp[(size_t)r * C_IN + 16 + m];
      float ss = v0 * v0 + v1 * v1;
      ss += __shfl_xor(ss, 1, 32);
      ss += __shfl_xor(ss, 2, 32);
      ss += __shfl_xor(ss, 4, 32);
      ss += __shfl_xor(ss, 8, 32);
      float inv = rsqrtf(ss * (1.0f / 32.0f) + 1e-5f);
      op[(size_t)r * C_IN + m]      = v0 * inv * lw0;
      op[(size_t)r * C_IN + 16 + m] = v1 * inv * lw1;
    }
  } else {
#pragma unroll
    for (int r = 0; r < 8; ++r) {
      int nrow = rowbase + r;
      int cr = (nrow < N) ? nrow : N - 1;
      float v0 = c0[r] + x[(size_t)cr * C_IN + m];
      float v1 = c1[r] + x[(size_t)cr * C_IN + 16 + m];
      float ss = v0 * v0 + v1 * v1;
      ss += __shfl_xor(ss, 1, 32);
      ss += __shfl_xor(ss, 2, 32);
      ss += __shfl_xor(ss, 4, 32);
      ss += __shfl_xor(ss, 8, 32);
      float inv = rsqrtf(ss * (1.0f / 32.0f) + 1e-5f);
      if (nrow < N) {
        out[(size_t)nrow * C_IN + m]      = v0 * inv * lw0;
        out[(size_t)nrow * C_IN + 16 + m] = v1 * inv * lw1;
      }
    }
  }
}

// ---------------------------------------------------------------------------
extern "C" void kernel_launch(void* const* d_in, const int* in_sizes, int n_in,
                              void* d_out, int out_size, void* d_ws, size_t ws_size,
                              hipStream_t stream) {
  const float* x    = (const float*)d_in[0];
  const float* ea   = (const float*)d_in[1];
  const float* Wee  = (const float*)d_in[2];
  const float* Wq   = (const float*)d_in[3];
  const float* Wk   = (const float*)d_in[4];
  const float* Wv   = (const float*)d_in[5];
  const float* Wle  = (const float*)d_in[6];
  const float* Wsk  = (const float*)d_in[7];
  const float* Wout = (const float*)d_in[8];
  const float* lnw  = (const float*)d_in[9];
  const int*   ei   = (const int*)d_in[10];
  float* out = (float*)d_out;

  const int N = in_sizes[0] / C_IN;   // 50000
  const int E = in_sizes[1] / 2;      // 800000

  // Carve workspace (total ~117 MB)
  char* ws = (char*)d_ws;
  size_t off = 0;
  auto carve = [&](size_t bytes) {
    char* ptr = ws + off;
    off += (bytes + 255) & ~(size_t)255;
    return ptr;
  };
  float*    qkv    = (float*)carve((size_t)N * QKV_STRIDE * sizeof(float));
  float*    agg    = (float*)carve((size_t)N * HD * sizeof(float));
  float*    pbuf   = (float*)carve((size_t)E * 4 * sizeof(float));
  unsigned* segmax = (unsigned*)carve((size_t)N * 4 * sizeof(unsigned));
  float*    segsum = (float*)carve((size_t)N * 4 * sizeof(float));
  float*    Wc     = (float*)carve(HD * 2 * sizeof(float));

  hipMemsetAsync(segmax, 0, (size_t)N * 4 * sizeof(unsigned), stream);
  hipMemsetAsync(segsum, 0, (size_t)N * 4 * sizeof(float), stream);

  wc_kernel<<<1, 256, 0, stream>>>(Wle, Wee, Wc);

  const int ntiles = (N + 15) / 16;
  proj_kernel<<<ntiles, 128, 0, stream>>>(x, Wq, Wk, Wv, Wsk, qkv, agg, N);

  const int t1 = E * 4;
  edge_logits_kernel<<<(t1 + 255) / 256, 256, 0, stream>>>(qkv, ea, Wc, ei,
                                                           pbuf, segmax, E);
  edge_softmax_kernel<<<(t1 + 255) / 256, 256, 0, stream>>>(ei, segmax, pbuf,
                                                            segsum, E);
  const long t3 = (long)E * 32;
  edge_aggregate_kernel<<<(int)((t3 + 255) / 256), 256, 0, stream>>>(
      qkv, ea, Wc, ei, pbuf, segsum, agg, E);

  out_kernel<<<(ntiles + 3) / 4, 128, 0, stream>>>(agg, x, Wout, lnw, out, N);
}
